// CCL_Module_48155173323002
// MI455X (gfx1250) — matrix operand order repaired and checked by reading, not compiled, not run on hardware
//
#include <hip/hip_runtime.h>

typedef __attribute__((ext_vector_type(16))) _Float16 v16h;
typedef __attribute__((ext_vector_type(8)))  float    v8f;

#define Cc 64
#define Wd 128
#define Hd 128
#define Bd 8
#define HWs (Hd * Wd)
#define STRIDE 70      // f16 per staged position (64 + 6 pad -> 35 dwords, odd stride)
#define NPOS_W 131     // staged warp positions: w = -1 .. 129
#define GROW 20        // padded dwords per Gram row (16 + 4; halves hit disjoint banks)
#define GSZ  (16 * GROW)

__launch_bounds__(256)
__global__ void ccl_flow_kernel(const float* __restrict__ f1,
                                const float* __restrict__ f2,
                                float* __restrict__ out) {
  __shared__ float    sScale1[Wd];
  __shared__ float    sScaleW[3][Wd];
  __shared__ _Float16 sA[Wd * STRIDE];
  __shared__ _Float16 sW[3][NPOS_W * STRIDE];
  __shared__ float    sG[8 * 6 * GSZ];   // [tile][dh][g][m][n] full Gram dumps

  const int tid = threadIdx.x;
  const int b   = blockIdx.x / Hd;
  const int h   = blockIdx.x % Hd;
  const size_t fbase = (size_t)b * Cc * HWs;

  // ---------------- Pass 1: per-position L2 norms ----------------
  // 512 tasks: row 0 = feature1 @ h ; rows 1..3 = feature2 @ h-1,h,h+1
  for (int task = tid; task < 512; task += 256) {
    const int row = task >> 7;
    const int w   = task & 127;
    float s = 0.f;
    if (row == 0) {
      const float* p = f1 + fbase + (size_t)h * Wd + w;
      #pragma unroll 8
      for (int c = 0; c < Cc; ++c) { float x = p[(size_t)c * HWs]; s += x * x; }
      sScale1[w] = 1.f / fmaxf(sqrtf(s), 1e-12f);
    } else {
      const int hh = h + row - 2;
      if (hh >= 0 && hh < Hd) {
        const float* p = f2 + fbase + (size_t)hh * Wd + w;
        #pragma unroll 8
        for (int c = 0; c < Cc; ++c) { float x = p[(size_t)c * HWs]; s += x * x; }
        sScaleW[row - 1][w] = 1.f / fmaxf(sqrtf(s), 1e-12f);
      } else {
        sScaleW[row - 1][w] = 0.f;
      }
    }
  }
  __syncthreads();

  // ---------------- Pass 2: stage normalized f16 tiles ----------------
  // feature1 row (position-major: sA[w*STRIDE + c])
  for (int idx = tid; idx < Cc * Wd; idx += 256) {
    const int c = idx >> 7;
    const int w = idx & 127;
    const float x = f1[fbase + (size_t)c * HWs + (size_t)h * Wd + w] * sScale1[w];
    sA[w * STRIDE + c] = (_Float16)x;
  }
  // feature2 rows h-1..h+1, staged at index (w+1); zero pads for w=-1,128,129
  for (int r = 0; r < 3; ++r) {
    const int hh = h + r - 1;
    const bool ok = (hh >= 0 && hh < Hd);
    const size_t base2 = fbase + (size_t)(ok ? hh : 0) * Wd;
    for (int idx = tid; idx < Cc * Wd; idx += 256) {
      const int c = idx >> 7;
      const int w = idx & 127;
      float x = 0.f;
      if (ok) x = f2[base2 + (size_t)c * HWs + w] * sScaleW[r][w];
      sW[r][(w + 1) * STRIDE + c] = (_Float16)x;
    }
    for (int idx = tid; idx < 3 * Cc; idx += 256) {
      const int which = idx >> 6;                         // 0,1,2
      const int c     = idx & 63;
      const int pi    = (which == 0) ? 0 : (128 + which); // 0, 129, 130
      sW[r][pi * STRIDE + c] = (_Float16)0.f;
    }
  }
  __syncthreads();

  // ---------------- WMMA: banded Gram correlation ----------------
  const int wv   = tid >> 5;      // wave id = w-tile id (8 waves, 8 tiles)
  const int lane = tid & 31;
  const int lh   = lane >> 4;     // lane half
  const int n    = lane & 15;     // column / M index
  const int w0   = wv << 4;

  union { v16h v; unsigned u[8]; } A0, A1, B0, B1;

  // A fragments: 16x32 f16, M = n, K layout per ISA 7.12.2
  {
    const _Float16* ap = sA + (w0 + n) * STRIDE;
    #pragma unroll
    for (int vv = 0; vv < 8; ++vv) {
      const int ch = ((vv >= 4) ? 16 : 0) + lh * 8 + (vv & 3) * 2;
      A0.u[vv] = *(const unsigned*)(ap + ch);
      A1.u[vv] = *(const unsigned*)(ap + 32 + ch);
    }
  }

  // per-lane Gram dump base: element (m = r+8*lh, n)
  float* gbaseLane = sG + lh * (8 * GROW) + n;

  #pragma unroll
  for (int dh = 0; dh < 3; ++dh) {
    #pragma unroll
    for (int g = 0; g < 2; ++g) {
      const int off = (g == 0) ? -1 : 2;   // B column tile shift
      // B fragments: 32x16 f16, N = n, VGPR v holds K = {(v>=4?16:0)+8*lh+2*(v&3), +1}
      const _Float16* bp = sW[dh] + (w0 + off + n + 1) * STRIDE;
      #pragma unroll
      for (int vv = 0; vv < 8; ++vv) {
        const int ch = ((vv >= 4) ? 16 : 0) + lh * 8 + (vv & 3) * 2;
        B0.u[vv] = *(const unsigned*)(bp + ch);
        B1.u[vv] = *(const unsigned*)(bp + 32 + ch);
      }
      v8f acc = {};
      acc = __builtin_amdgcn_wmma_f32_16x16x32_f16(false, A0.v, false, B0.v,
                                                   (short)0, acc, false, false);
      acc = __builtin_amdgcn_wmma_f32_16x16x32_f16(false, A1.v, false, B1.v,
                                                   (short)0, acc, false, false);
      // Unconditional full Gram dump: D[m][n] = corr(w0+m, w0+off+n)
      float* gp = gbaseLane + ((wv * 6 + dh * 2 + g) * GSZ);
      #pragma unroll
      for (int r = 0; r < 8; ++r)
        gp[r * GROW] = acc[r];
    }
  }
  __syncthreads();

  // ---------------- Band pick + softmax + expected displacement ----------------
  if (tid < Wd) {
    const int w    = tid;
    const int tile = w >> 4;
    const int i    = w & 15;
    float vals[9], mx = -1e30f;
    #pragma unroll
    for (int k = 0; k < 9; ++k) {
      const int dh  = k / 3;
      const int dwi = k % 3;
      const int j1  = i + dwi;          // column in Gram g=0 (B shift -1)
      const int g   = (j1 < 16) ? 0 : 1;
      const int j   = (j1 < 16) ? j1 : (j1 - 3);  // Gram g=1 has B shift +2
      vals[k] = sG[(tile * 6 + dh * 2 + g) * GSZ + i * GROW + j] * 10.0f;
      mx = fmaxf(mx, vals[k]);
    }
    float sum = 0.f, fw = 0.f, fh = 0.f;
    #pragma unroll
    for (int k = 0; k < 9; ++k) {
      const float e = __expf(vals[k] - mx);
      sum += e;
      fw  += e * (float)((k % 3) - 1);
      fh  += e * (float)((k / 3) - 1);
    }
    const float inv = 1.f / sum;
    const size_t obase = ((size_t)b * 2) * HWs + (size_t)h * Wd + w;
    out[obase]       = fw * inv;   // flow_w (channel 0)
    out[obase + HWs] = fh * inv;   // flow_h (channel 1)
  }
}

extern "C" void kernel_launch(void* const* d_in, const int* in_sizes, int n_in,
                              void* d_out, int out_size, void* d_ws, size_t ws_size,
                              hipStream_t stream) {
  const float* f1 = (const float*)d_in[0];
  const float* f2 = (const float*)d_in[1];
  float* out = (float*)d_out;
  ccl_flow_kernel<<<dim3(Bd * Hd), dim3(256), 0, stream>>>(f1, f2, out);
}
